// GQAReference_42958262895378
// MI455X (gfx1250) — matrix operand order repaired and checked by reading, not compiled, not run on hardware
//
#include <hip/hip_runtime.h>

// ---------------- problem constants (B=1) ----------------
#define S_LEN 2048
#define H_DIM 4096
#define NQ_H  32
#define NKV_H 8
#define HD    128
#define EQ    (NQ_H * HD)   // 4096
#define EK    (NKV_H * HD)  // 1024
#define SCALE_ATT 0.08838834764831845f  // 1/sqrt(128)
#define SOFTCAP_V 30.0f
#define RMS_EPS   1e-5f

typedef __attribute__((ext_vector_type(16))) __bf16 v16bf;
typedef __attribute__((ext_vector_type(8)))  __bf16 v8bf;
typedef __attribute__((ext_vector_type(8)))  float  v8f;

// ---------------- scalar bf16 helpers (bit-exact RNE, no header deps) ----------------
__device__ __forceinline__ __bf16 f2bf(float f) {
    union { float f; unsigned u; } in; in.f = f;
    unsigned u = in.u;
    u += 0x7fffu + ((u >> 16) & 1u);               // round to nearest even
    union { unsigned short s; __bf16 b; } out; out.s = (unsigned short)(u >> 16);
    return out.b;
}
__device__ __forceinline__ float bf2f(__bf16 b) {
    union { unsigned short s; __bf16 b; } in; in.b = b;
    union { unsigned u; float f; } out; out.u = ((unsigned)in.s) << 16;
    return out.f;
}
__device__ __forceinline__ float fast_tanh(float x) {
    float xc = fminf(fmaxf(x, -15.f), 15.f);
    float e  = __expf(2.f * xc);
    return (e - 1.f) / (e + 1.f);
}

// ---- CDNA5 async global->LDS copy (ASYNCcnt-tracked DMA into LDS) ----
// 16B per lane; LDS dest offset in a VGPR, 64-bit global address in a VGPR pair.
#define ASYNC_CP16(ldsOff, gptr)                                           \
    asm volatile("global_load_async_to_lds_b128 %0, %1, off"               \
                 :: "v"(ldsOff), "v"(gptr) : "memory")
#define ASYNC_WAIT() asm volatile("s_wait_asynccnt 0x0" ::: "memory")

// =====================================================================
// Generic NT bf16 WMMA GEMM:  C[M,N] = A[M,K] * B[N,K]^T
//   A row-major (M,K) lda ; B row-major (N,K) ldb  (K contiguous both)
//   MODE 0: f32 out (+ optional residual add)
//   MODE 1: bf16 out, attention-score epilogue (scale + tanh softcap)
//   MODE 2: bf16 out
//   cmode 1: skip blocks entirely above the causal diagonal (scores)
//   cmode 2: limit K loop to causal extent of the block's rows (PV)
// Wave tile 32x64, 4 waves stacked in M -> 128x64 block tile.
// A/B tiles staged to LDS via async DMA, double buffered; fragments via
// ds_load_b128 from bank-padded rows (stride 80B = 20 banks, conflict-free).
// z-batching: A += z*sAz, B += (z/bGroup)*sBz, C flat offset z*sCz (GQA).
// =====================================================================
union ABfrag { v16bf v; v8bf h[2]; };

#define PR 40  // padded LDS row stride in bf16 elements (80 bytes)

template <int MODE>
__global__ __launch_bounds__(128)
void gemm_bf16_nt(const __bf16* __restrict__ A, const __bf16* __restrict__ B,
                  float* __restrict__ Cf, __bf16* __restrict__ Cb,
                  const float* __restrict__ res,
                  int M, int N, int K, int lda, int ldb, int ldc,
                  long sAz, long sBz, long sCz, int bGroup, int cmode)
{
    __shared__ __align__(16) __bf16 sA[2][128 * PR];
    __shared__ __align__(16) __bf16 sB[2][64 * PR];

    const int z = blockIdx.z;
    A += (long)z * sAz;
    B += (long)(z / bGroup) * sBz;
    const long coff = (long)z * sCz;

    const int tid  = threadIdx.x;
    const int lane = tid & 31;
    const int wave = tid >> 5;
    const int hf   = lane >> 4;      // lane-group (0: lanes 0-15, 1: lanes 16-31)
    const int lr   = lane & 15;

    const int mbB = blockIdx.y * 128;          // block row base
    const int mb  = mbB + wave * 32;           // this wave's 32 output rows
    const int nb  = blockIdx.x * 64;           // block's 64 output cols

    // causal: whole block above diagonal -> nothing survives softmax, skip it.
    if (cmode == 1 && nb >= mbB + 128) return;
    // causal: probs are exact zeros for j > row; block rows end at mbB+127.
    int kEnd = K;
    if (cmode == 2) kEnd = min(K, mbB + 128);

    const __bf16* Ab = A + (long)mbB * lda;
    const __bf16* Bb = B + (long)nb * ldb;

    // stage one 128x32 A tile (8KB) + 64x32 B tile (4KB) into LDS buffer `buf`
    auto stage = [&](int buf, int k0) {
#pragma unroll
        for (int i = 0; i < 4; ++i) {          // A: 512 16B-chunks / 128 thr
            const int it = tid + i * 128;
            const int r = it >> 2, c = it & 3;
            const unsigned lo = (unsigned)(unsigned long long)(&sA[buf][r * PR + c * 8]);
            const __bf16* g = Ab + (long)r * lda + k0 + c * 8;
            ASYNC_CP16(lo, g);
            if (i == 0) __builtin_prefetch(g + 64, 0, 1);   // pull k0+64 toward L2
        }
#pragma unroll
        for (int i = 0; i < 2; ++i) {          // B: 256 16B-chunks / 128 thr
            const int it = tid + i * 128;
            const int r = it >> 2, c = it & 3;
            const unsigned lo = (unsigned)(unsigned long long)(&sB[buf][r * PR + c * 8]);
            const __bf16* g = Bb + (long)r * ldb + k0 + c * 8;
            ASYNC_CP16(lo, g);
            if (i == 0) __builtin_prefetch(g + 64, 0, 1);
        }
    };

    v8f acc[2][4];
#pragma unroll
    for (int i = 0; i < 2; ++i)
#pragma unroll
        for (int j = 0; j < 4; ++j)
#pragma unroll
            for (int r = 0; r < 8; ++r) acc[i][j][r] = 0.f;

    int buf = 0;
    stage(0, 0);
    ASYNC_WAIT();
    __syncthreads();

    for (int k0 = 0; k0 < kEnd; k0 += 32) {
        const int nxt = k0 + 32;
        if (nxt < kEnd) stage(buf ^ 1, nxt);   // DMA next tile during compute

        const __bf16* sAb = sA[buf];
        const __bf16* sBb = sB[buf];
        ABfrag a0, a1, bfr[4];
        // A frag (16x32): lane = row, K chunks {0..7,16..23} + 8*hf
        const int ar0 = (wave * 32 + lr) * PR + hf * 8;
        a0.h[0] = *(const v8bf*)(sAb + ar0);
        a0.h[1] = *(const v8bf*)(sAb + ar0 + 16);
        const int ar1 = ar0 + 16 * PR;
        a1.h[0] = *(const v8bf*)(sAb + ar1);
        a1.h[1] = *(const v8bf*)(sAb + ar1 + 16);
        // B frag (32x16): lane = col, K chunks {0..7,16..23} + 8*hf
#pragma unroll
        for (int j = 0; j < 4; ++j) {
            const int br = (16 * j + lr) * PR + hf * 8;
            bfr[j].h[0] = *(const v8bf*)(sBb + br);
            bfr[j].h[1] = *(const v8bf*)(sBb + br + 16);
        }
#pragma unroll
        for (int j = 0; j < 4; ++j) {
            acc[0][j] = __builtin_amdgcn_wmma_f32_16x16x32_bf16(
                false, a0.v, false, bfr[j].v, (short)0, acc[0][j], false, false);
            acc[1][j] = __builtin_amdgcn_wmma_f32_16x16x32_bf16(
                false, a1.v, false, bfr[j].v, (short)0, acc[1][j], false, false);
        }

        ASYNC_WAIT();       // next tile landed
        __syncthreads();    // all waves done reading `buf`, staging published
        buf ^= 1;
    }

    // D layout: vgpr r, lanes 0-15 -> (m = r, n = lane); lanes 16-31 -> (m = r+8)
#pragma unroll
    for (int i = 0; i < 2; ++i)
#pragma unroll
        for (int j = 0; j < 4; ++j)
#pragma unroll
            for (int r = 0; r < 8; ++r) {
                const int m = mb + 16 * i + r + 8 * hf;
                const int n = nb + 16 * j + lr;
                const long idx = coff + (long)m * ldc + n;
                float val = acc[i][j][r];
                if constexpr (MODE == 0) {
                    if (res) val += res[idx];
                    Cf[idx] = val;
                } else if constexpr (MODE == 1) {
                    float t = val * SCALE_ATT * (1.0f / SOFTCAP_V);
                    Cb[idx] = f2bf(SOFTCAP_V * fast_tanh(t));
                } else {
                    Cb[idx] = f2bf(val);
                }
            }
}

// ---------------- pre-attention RMSNorm (fp32 in, bf16 out) ----------------
__global__ __launch_bounds__(256)
void pre_rmsnorm(const float* __restrict__ x, const float* __restrict__ w,
                 __bf16* __restrict__ out)
{
    const int s = blockIdx.x;
    const float* row = x + (long)s * H_DIM;
    float ss = 0.f;
    for (int j = threadIdx.x; j < H_DIM; j += 256) { float v = row[j]; ss += v * v; }
    __shared__ float red[256];
    red[threadIdx.x] = ss;
    __syncthreads();
    for (int st = 128; st > 0; st >>= 1) {
        if (threadIdx.x < st) red[threadIdx.x] += red[threadIdx.x + st];
        __syncthreads();
    }
    const float inv = rsqrtf(red[0] / (float)H_DIM + RMS_EPS);
    for (int j = threadIdx.x; j < H_DIM; j += 256)
        out[(long)s * H_DIM + j] = f2bf(w[j] * row[j] * inv);
}

// ---------------- fp32 -> bf16 weight conversion ----------------
__global__ void cvt_f32_bf16(const float* __restrict__ in, __bf16* __restrict__ out, long n)
{
    long i = (long)blockIdx.x * blockDim.x + threadIdx.x;
    const long stride = (long)gridDim.x * blockDim.x;
    for (; i < n; i += stride) out[i] = f2bf(in[i]);
}

// ---------------- per-head RMSNorm + RoPE, repack to [head][S][D] bf16 ----------------
__global__ __launch_bounds__(128)
void qk_norm_rope(const float* __restrict__ src, int E,
                  const float* __restrict__ cosT, const float* __restrict__ sinT,
                  const float* __restrict__ w, __bf16* __restrict__ dst)
{
    const int s  = blockIdx.x;
    const int hd = blockIdx.y;
    const int d  = threadIdx.x;  // 0..127
    const float x = src[(long)s * E + hd * HD + d];
    __shared__ float red[128];
    __shared__ float xn[128];
    red[d] = x * x;
    __syncthreads();
    for (int st = 64; st > 0; st >>= 1) {
        if (d < st) red[d] += red[d + st];
        __syncthreads();
    }
    const float inv = rsqrtf(red[0] / (float)HD + RMS_EPS);
    const float nv  = w[d] * x * inv;
    xn[d] = nv;
    __syncthreads();
    const float partner = (d < 64) ? -xn[d + 64] : xn[d - 64];  // rotate_half
    const float c  = cosT[(long)s * HD + d];
    const float sn = sinT[(long)s * HD + d];
    dst[((long)hd * S_LEN + s) * HD + d] = f2bf(nv * c + partner * sn);
}

// ---------------- V: fp32 (S, NKV*D) -> bf16 transposed [kvh][D][S] ----------------
__global__ void v_transpose(const float* __restrict__ v, __bf16* __restrict__ vt)
{
    const long n = (long)S_LEN * EK;
    for (long i = (long)blockIdx.x * blockDim.x + threadIdx.x; i < n;
         i += (long)gridDim.x * blockDim.x) {
        const int s = (int)(i / EK);
        const int e = (int)(i % EK);   // e = kvh*D + d
        vt[(long)e * S_LEN + s] = f2bf(v[i]);
    }
}

// ---------------- causal softmax (in-place, bf16 scores -> bf16 probs) ----------------
__global__ __launch_bounds__(256)
void softmax_causal(__bf16* __restrict__ pbuf)
{
    const int i = blockIdx.x;      // query row
    const int h = blockIdx.y;      // q head
    __bf16* row = pbuf + ((long)h * S_LEN + i) * S_LEN;
    const int L = i + 1;           // valid (causal) length
    __shared__ float red[256];

    float mx = -3.0e38f;
    for (int j = threadIdx.x; j < L; j += 256) mx = fmaxf(mx, bf2f(row[j]));
    red[threadIdx.x] = mx;
    __syncthreads();
    for (int st = 128; st > 0; st >>= 1) {
        if (threadIdx.x < st) red[threadIdx.x] = fmaxf(red[threadIdx.x], red[threadIdx.x + st]);
        __syncthreads();
    }
    const float rowmax = red[0];
    __syncthreads();

    float sum = 0.f;
    for (int j = threadIdx.x; j < L; j += 256) sum += __expf(bf2f(row[j]) - rowmax);
    red[threadIdx.x] = sum;
    __syncthreads();
    for (int st = 128; st > 0; st >>= 1) {
        if (threadIdx.x < st) red[threadIdx.x] += red[threadIdx.x + st];
        __syncthreads();
    }
    const float inv = 1.f / red[0];

    for (int j = threadIdx.x; j < S_LEN; j += 256) {
        const float p = (j < L) ? __expf(bf2f(row[j]) - rowmax) * inv : 0.f;
        row[j] = f2bf(p);
    }
}

// =====================================================================
// Orchestration
// =====================================================================
extern "C" void kernel_launch(void* const* d_in, const int* in_sizes, int n_in,
                              void* d_out, int out_size, void* d_ws, size_t ws_size,
                              hipStream_t stream)
{
    (void)in_sizes; (void)n_in; (void)out_size; (void)ws_size;
    const float* hidden = (const float*)d_in[0];
    // d_in[1] = attention_mask (pure causal -> applied analytically, unused)
    const float* cosT   = (const float*)d_in[2];
    const float* sinT   = (const float*)d_in[3];
    const float* pre_w  = (const float*)d_in[4];
    const float* wq     = (const float*)d_in[5];
    const float* wk     = (const float*)d_in[6];
    const float* wv     = (const float*)d_in[7];
    const float* wo     = (const float*)d_in[8];
    const float* qn_w   = (const float*)d_in[9];
    const float* kn_w   = (const float*)d_in[10];
    float* out = (float*)d_out;

    // ---- workspace carve-up (bytes, 256-aligned). Total ~445 MB. ----
    char* ws = (char*)d_ws;
    size_t off = 0;
    auto carve = [&](size_t bytes) -> char* {
        char* p = ws + off;
        off += (bytes + 255) & ~(size_t)255;
        return p;
    };
    __bf16* hb  = (__bf16*)carve((size_t)S_LEN * H_DIM * 2);       // normed hidden, bf16
    __bf16* wqb = (__bf16*)carve((size_t)EQ * H_DIM * 2);
    __bf16* wkb = (__bf16*)carve((size_t)EK * H_DIM * 2);
    __bf16* wvb = (__bf16*)carve((size_t)EK * H_DIM * 2);
    __bf16* wob = (__bf16*)carve((size_t)H_DIM * EQ * 2);
    float*  tmp = (float*) carve((size_t)S_LEN * EQ * 4);          // raw QKV proj (f32), reused
    __bf16* qb  = (__bf16*)carve((size_t)NQ_H * S_LEN * HD * 2);   // [qh][S][D]
    __bf16* kb  = (__bf16*)carve((size_t)NKV_H * S_LEN * HD * 2);  // [kvh][S][D]
    __bf16* vtb = (__bf16*)carve((size_t)NKV_H * HD * S_LEN * 2);  // [kvh][D][S]
    __bf16* pb  = (__bf16*)carve((size_t)NQ_H * S_LEN * S_LEN * 2);// scores/probs, L2-resident
    __bf16* ab  = (__bf16*)carve((size_t)S_LEN * EQ * 2);          // attn context [S][NQ*D]

    // 1) pre-RMSNorm -> bf16
    pre_rmsnorm<<<dim3(S_LEN), dim3(256), 0, stream>>>(hidden, pre_w, hb);

    // 2) weight conversion fp32 -> bf16 (stays hot in 192MB L2 afterwards)
    cvt_f32_bf16<<<dim3(2048), dim3(256), 0, stream>>>(wq, wqb, (long)EQ * H_DIM);
    cvt_f32_bf16<<<dim3(1024), dim3(256), 0, stream>>>(wk, wkb, (long)EK * H_DIM);
    cvt_f32_bf16<<<dim3(1024), dim3(256), 0, stream>>>(wv, wvb, (long)EK * H_DIM);
    cvt_f32_bf16<<<dim3(2048), dim3(256), 0, stream>>>(wo, wob, (long)H_DIM * EQ);

    // 3) Q = h @ wq^T   (M=S, N=EQ, K=H)
    gemm_bf16_nt<0><<<dim3(EQ / 64, S_LEN / 128, 1), dim3(128), 0, stream>>>(
        hb, wqb, tmp, nullptr, nullptr, S_LEN, EQ, H_DIM, H_DIM, H_DIM, EQ, 0, 0, 0, 1, 0);
    qk_norm_rope<<<dim3(S_LEN, NQ_H), dim3(128), 0, stream>>>(tmp, EQ, cosT, sinT, qn_w, qb);

    // 4) K = h @ wk^T   (M=S, N=EK, K=H)
    gemm_bf16_nt<0><<<dim3(EK / 64, S_LEN / 128, 1), dim3(128), 0, stream>>>(
        hb, wkb, tmp, nullptr, nullptr, S_LEN, EK, H_DIM, H_DIM, H_DIM, EK, 0, 0, 0, 1, 0);
    qk_norm_rope<<<dim3(S_LEN, NKV_H), dim3(128), 0, stream>>>(tmp, EK, cosT, sinT, kn_w, kb);

    // 5) V = h @ wv^T, stored transposed per kv head -> PV GEMM is NT too
    gemm_bf16_nt<0><<<dim3(EK / 64, S_LEN / 128, 1), dim3(128), 0, stream>>>(
        hb, wvb, tmp, nullptr, nullptr, S_LEN, EK, H_DIM, H_DIM, H_DIM, EK, 0, 0, 0, 1, 0);
    v_transpose<<<dim3(2048), dim3(256), 0, stream>>>(tmp, vtb);

    // 6) scores[h] = softcap(Q_h K_{h/4}^T * scale), 32 heads in grid.z,
    //    blocks above the causal diagonal skipped (cmode=1)
    gemm_bf16_nt<1><<<dim3(S_LEN / 64, S_LEN / 128, NQ_H), dim3(128), 0, stream>>>(
        qb, kb, nullptr, pb, nullptr, S_LEN, S_LEN, HD, HD, HD, S_LEN,
        (long)S_LEN * HD, (long)S_LEN * HD, (long)S_LEN * S_LEN, NQ_H / NKV_H, 1);

    // 7) causal softmax in-place (fp32 math, bf16 storage; also zeroes j>i)
    softmax_causal<<<dim3(S_LEN, NQ_H), dim3(256), 0, stream>>>(pb);

    // 8) context[h] = P_h @ V_{h/4}  (M=S, N=D, K limited causally, cmode=2)
    gemm_bf16_nt<2><<<dim3(HD / 64, S_LEN / 128, NQ_H), dim3(128), 0, stream>>>(
        pb, vtb, nullptr, ab, nullptr, S_LEN, HD, S_LEN, S_LEN, S_LEN, EQ,
        (long)S_LEN * S_LEN, (long)HD * S_LEN, (long)HD, NQ_H / NKV_H, 2);

    // 9) out = residual + context @ wo^T  (M=S, N=H, K=EQ)
    gemm_bf16_nt<0><<<dim3(H_DIM / 64, S_LEN / 128, 1), dim3(128), 0, stream>>>(
        ab, wob, out, nullptr, hidden, S_LEN, H_DIM, EQ, EQ, EQ, H_DIM, 0, 0, 0, 1, 0);
}
